// MultiHeadTokenAttention_1125281432101
// MI455X (gfx1250) — matrix operand order repaired and checked by reading, not compiled, not run on hardware
//
#include <hip/hip_runtime.h>
#include <hip/hip_bf16.h>

// ---------------------------------------------------------------------------
// MultiHeadTokenAttention for MI455X (gfx1250, wave32, WMMA)
//   B=8, Q=32, S=64, T=128, H=1024, HEADS=16, HEAD_DIM=64
// Memory-bound (~0.65 GB traffic after KV-fusion vs 317 GFLOP):
//   - K and V projections FUSED (ini_k, the 268 MB input, streamed ONCE)
//   - fp32 WMMA 16x16x4 for the deep projections (exact precision; confirmed
//     selected by the compiler), batched fragment loads so the load clause
//     overlaps the WMMA chain instead of s_wait_loadcnt 0 per instruction
//   - q/k/v stored f16, QK^T scores on v_wmma_f32_16x16x32_f16, alpha@V on VALU FMA
//   - scores/alpha live only in LDS (134 MB never touches HBM)
//   - O-projection writes d_out, LayerNorm runs in place
// ---------------------------------------------------------------------------

typedef __attribute__((ext_vector_type(2)))  float    v2f;
typedef __attribute__((ext_vector_type(8)))  float    v8f;
typedef __attribute__((ext_vector_type(16))) _Float16 v16h;

#define HH     1024
#define NB     8
#define NQ     32
#define NS     64
#define NT     128
#define NHEADS 16
#define HD     64

#if __has_builtin(__builtin_amdgcn_wmma_f32_16x16x4_f32)
#define USE_WMMA_F32 1
#pragma message("CDNA5: projections use V_WMMA_F32_16X16X4_F32 (fp32 path)")
#else
#define USE_WMMA_F32 0
#pragma message("CDNA5: f32 WMMA builtin unavailable; projections use f16 WMMA fallback")
#endif

// ---- fragment load helpers -------------------------------------------------

__device__ __forceinline__ v16h ldA_f16(const _Float16* __restrict__ p) {
  v16h r;
#pragma unroll
  for (int i = 0; i < 8; ++i) { r[i] = p[i]; r[8 + i] = p[16 + i]; }
  return r;
}
__device__ __forceinline__ v16h ldB_f16(const _Float16* __restrict__ p) {
  v16h r;
#pragma unroll
  for (int i = 0; i < 16; ++i) r[i] = p[i];
  return r;
}
__device__ __forceinline__ v16h cvtA_f16(const float* __restrict__ p) {
  v16h r;
#pragma unroll
  for (int i = 0; i < 8; ++i) { r[i] = (_Float16)p[i]; r[8 + i] = (_Float16)p[16 + i]; }
  return r;
}
__device__ __forceinline__ v16h cvtB_f16(const float* __restrict__ p) {
  v16h r;
#pragma unroll
  for (int i = 0; i < 16; ++i) r[i] = (_Float16)p[i];
  return r;
}

// ---------------------------------------------------------------------------
// Generic GEMM: Y[n, j] = sum_d X[n,d] * W[j,d] + bias[j]   (K = cols = 1024)
// block = 128 threads (4 waves); wave -> 16 rows x 64 cols (4 accumulators)
// Loads batched ahead of the WMMA chain to overlap fetch with the matrix pipe.
// ---------------------------------------------------------------------------
template <typename OutT>
__global__ __launch_bounds__(128) void gemm_bias_kernel(
    const float* __restrict__ X, const float* __restrict__ W,
    const float* __restrict__ bias, OutT* __restrict__ Y) {
  const int lane = threadIdx.x & 31;
  const int wave = threadIdx.x >> 5;
  const int m0   = blockIdx.y * 64 + wave * 16;
  const int j0   = blockIdx.x * 64;
  const int row  = lane & 15;
  const int kh   = lane >> 4;

  v8f acc[4];
#pragma unroll
  for (int j = 0; j < 4; ++j) acc[j] = {};

#if USE_WMMA_F32
  const float* xp = X + (size_t)(m0 + row) * HH + 2 * kh;
  const float* wp = W + (size_t)(j0 + row) * HH + 2 * kh;
  for (int k = 0; k < HH; k += 4) {
    v2f a = *(const v2f*)(xp + k);
    v2f bf[4];
#pragma unroll
    for (int j = 0; j < 4; ++j) bf[j] = *(const v2f*)(wp + (size_t)j * 16 * HH + k);
#pragma unroll
    for (int j = 0; j < 4; ++j)
      acc[j] = __builtin_amdgcn_wmma_f32_16x16x4_f32(
          false, a, false, bf[j], (short)0, acc[j], false, false);
  }
#else
  const float* xp = X + (size_t)(m0 + row) * HH + 8 * kh;
  const float* wp = W + (size_t)(j0 + row) * HH + 16 * kh;
  for (int k = 0; k < HH; k += 32) {
    v16h a = cvtA_f16(xp + k);
    v16h bf[4];
#pragma unroll
    for (int j = 0; j < 4; ++j) bf[j] = cvtB_f16(wp + (size_t)j * 16 * HH + k);
#pragma unroll
    for (int j = 0; j < 4; ++j)
      acc[j] = __builtin_amdgcn_wmma_f32_16x16x32_f16(
          false, a, false, bf[j], (short)0, acc[j], false, false);
  }
#endif

  const int n    = lane & 15;
  const int moff = kh * 8;
#pragma unroll
  for (int j = 0; j < 4; ++j) {
    const int   col = j0 + j * 16 + n;
    const float bj  = bias[col];
#pragma unroll
    for (int r = 0; r < 8; ++r) {
      Y[(size_t)(m0 + moff + r) * HH + col] = (OutT)(acc[j][r] + bj);
    }
  }
}

// ---------------------------------------------------------------------------
// FUSED K+V projection: streams X = ini_k (the 268 MB input) exactly once,
// sharing the A fragments between the Wk and Wv WMMA chains. All 8 B
// fragments are loaded as one clause before the 8-WMMA chain.
// ---------------------------------------------------------------------------
__global__ __launch_bounds__(128) void gemm_kv_kernel(
    const float* __restrict__ X,
    const float* __restrict__ Wk, const float* __restrict__ bk,
    const float* __restrict__ Wv, const float* __restrict__ bv,
    _Float16* __restrict__ K, _Float16* __restrict__ V) {
  const int lane = threadIdx.x & 31;
  const int wave = threadIdx.x >> 5;
  const int m0   = blockIdx.y * 64 + wave * 16;
  const int j0   = blockIdx.x * 64;
  const int row  = lane & 15;
  const int kh   = lane >> 4;

  v8f kacc[4], vacc[4];
#pragma unroll
  for (int j = 0; j < 4; ++j) { kacc[j] = {}; vacc[j] = {}; }

#if USE_WMMA_F32
  const float* xp  = X  + (size_t)(m0 + row) * HH + 2 * kh;
  const float* wkp = Wk + (size_t)(j0 + row) * HH + 2 * kh;
  const float* wvp = Wv + (size_t)(j0 + row) * HH + 2 * kh;
  for (int k = 0; k < HH; k += 4) {
    v2f a = *(const v2f*)(xp + k);
    v2f bkf[4], bvf[4];
#pragma unroll
    for (int j = 0; j < 4; ++j) {
      bkf[j] = *(const v2f*)(wkp + (size_t)j * 16 * HH + k);
      bvf[j] = *(const v2f*)(wvp + (size_t)j * 16 * HH + k);
    }
#pragma unroll
    for (int j = 0; j < 4; ++j) {
      kacc[j] = __builtin_amdgcn_wmma_f32_16x16x4_f32(
          false, a, false, bkf[j], (short)0, kacc[j], false, false);
      vacc[j] = __builtin_amdgcn_wmma_f32_16x16x4_f32(
          false, a, false, bvf[j], (short)0, vacc[j], false, false);
    }
  }
#else
  const float* xp  = X  + (size_t)(m0 + row) * HH + 8 * kh;
  const float* wkp = Wk + (size_t)(j0 + row) * HH + 16 * kh;
  const float* wvp = Wv + (size_t)(j0 + row) * HH + 16 * kh;
  for (int k = 0; k < HH; k += 32) {
    v16h a = cvtA_f16(xp + k);
    v16h bkf[4], bvf[4];
#pragma unroll
    for (int j = 0; j < 4; ++j) {
      bkf[j] = cvtB_f16(wkp + (size_t)j * 16 * HH + k);
      bvf[j] = cvtB_f16(wvp + (size_t)j * 16 * HH + k);
    }
#pragma unroll
    for (int j = 0; j < 4; ++j) {
      kacc[j] = __builtin_amdgcn_wmma_f32_16x16x32_f16(
          false, a, false, bkf[j], (short)0, kacc[j], false, false);
      vacc[j] = __builtin_amdgcn_wmma_f32_16x16x32_f16(
          false, a, false, bvf[j], (short)0, vacc[j], false, false);
    }
  }
#endif

  const int n    = lane & 15;
  const int moff = kh * 8;
#pragma unroll
  for (int j = 0; j < 4; ++j) {
    const int   col = j0 + j * 16 + n;
    const float bkj = bk[col];
    const float bvj = bv[col];
#pragma unroll
    for (int r = 0; r < 8; ++r) {
      const size_t idx = (size_t)(m0 + moff + r) * HH + col;
      K[idx] = (_Float16)(kacc[j][r] + bkj);
      V[idx] = (_Float16)(vacc[j][r] + bvj);
    }
  }
}

// ---------------------------------------------------------------------------
// Fused attention: one block (128 thr / 4 waves) per (b,s); wave w handles
// heads h = w, w+4, w+8, w+12.  scores = WMMA f16 (Q_h 32x64 * K_h^T 64x128)
// -> LDS (stride 132), masked softmax over T in LDS, alpha@V via VALU FMA,
// writes res[b,q,s,:] directly (scores/alpha never hit HBM).
// ---------------------------------------------------------------------------
__global__ __launch_bounds__(128) void attn_kernel(
    const _Float16* __restrict__ q16,  // (B*Q, H)
    const _Float16* __restrict__ k16,  // (B*S*T, H)
    const _Float16* __restrict__ v16,  // (B*S*T, H)
    const float* __restrict__ mask,    // (B, S, T)
    float* __restrict__ res)           // (B, Q, S, H)
{
  __shared__ float sc[4][NQ][NT + 4];  // +4 pad to avoid bank conflicts

  const int bs   = blockIdx.x;   // b*64 + s
  const int b    = bs >> 6;
  const int s    = bs & 63;
  const int lane = threadIdx.x & 31;
  const int wave = threadIdx.x >> 5;
  const int n    = lane & 15;
  const int kh   = lane >> 4;

  const size_t kvbase = (size_t)bs * NT * HH;
  const float* maskp  = mask + (size_t)bs * NT;

  for (int hi = 0; hi < 4; ++hi) {
    const int h = wave + 4 * hi;

    // --- A fragments of Q_h (2 M-tiles x 2 K-steps of 32) ---
    v16h aF[2][2];
#pragma unroll
    for (int mt = 0; mt < 2; ++mt)
#pragma unroll
      for (int ks = 0; ks < 2; ++ks) {
        const _Float16* p = q16 + (size_t)(b * NQ + mt * 16 + n) * HH +
                            h * HD + ks * 32 + 16 * kh;
        aF[mt][ks] = ldB_f16(p);
      }

    // --- scores: 8 N-tiles of 16 tokens each ---
    for (int nt = 0; nt < 8; ++nt) {
      const int t = nt * 16 + n;
      const _Float16* kp = k16 + kvbase + (size_t)t * HH + h * HD + 16 * kh;
      v16h b0 = ldB_f16(kp);        // d = 16*kh .. +15
      v16h b1 = ldB_f16(kp + 32);   // d = 32 + 16*kh .. +15
      v8f a0 = {}, a1 = {};
      a0 = __builtin_amdgcn_wmma_f32_16x16x32_f16(false, aF[0][0], false, b0,
                                                  (short)0, a0, false, false);
      a0 = __builtin_amdgcn_wmma_f32_16x16x32_f16(false, aF[0][1], false, b1,
                                                  (short)0, a0, false, false);
      a1 = __builtin_amdgcn_wmma_f32_16x16x32_f16(false, aF[1][0], false, b0,
                                                  (short)0, a1, false, false);
      a1 = __builtin_amdgcn_wmma_f32_16x16x32_f16(false, aF[1][1], false, b1,
                                                  (short)0, a1, false, false);
      const float mv = maskp[t] * -10000.0f;
#pragma unroll
      for (int r = 0; r < 8; ++r) {
        sc[wave][kh * 8 + r][t]      = a0[r] * 0.125f + mv;  // 1/sqrt(64)
        sc[wave][16 + kh * 8 + r][t] = a1[r] * 0.125f + mv;
      }
    }
    __syncthreads();

    // --- softmax over T: lane m owns row m (32 lanes == 32 q-rows) ---
    {
      float* rowp = &sc[wave][lane][0];
      float  mx   = -1e30f;
      for (int t = 0; t < NT; ++t) mx = fmaxf(mx, rowp[t]);
      float sum = 0.0f;
      for (int t = 0; t < NT; ++t) {
        float e = __expf(rowp[t] - mx);
        rowp[t] = e;
        sum += e;
      }
      const float inv = 1.0f / sum;
      for (int t = 0; t < NT; ++t) rowp[t] *= inv;
    }
    __syncthreads();

    // --- out_h = alpha (32x128) @ V_h (128x64); lane owns d=lane, lane+32 ---
    {
      float acc[NQ][2];
#pragma unroll
      for (int m = 0; m < NQ; ++m) { acc[m][0] = 0.0f; acc[m][1] = 0.0f; }
      const _Float16* vp = v16 + kvbase + h * HD;
      for (int t = 0; t < NT; ++t) {
        const float v0 = (float)vp[(size_t)t * HH + lane];
        const float v1 = (float)vp[(size_t)t * HH + lane + 32];
#pragma unroll
        for (int m = 0; m < NQ; ++m) {
          const float a = sc[wave][m][t];  // broadcast read
          acc[m][0] += a * v0;
          acc[m][1] += a * v1;
        }
      }
#pragma unroll
      for (int m = 0; m < NQ; ++m) {
        float* op = res + (size_t)((b * NQ + m) * NS + s) * HH + h * HD;
        op[lane]      = acc[m][0];
        op[lane + 32] = acc[m][1];
      }
    }
    __syncthreads();
  }
}

// ---------------------------------------------------------------------------
// In-place LayerNorm over last dim (1024) of d_out; one block per row.
// ---------------------------------------------------------------------------
__global__ __launch_bounds__(256) void layernorm_inplace(
    float* __restrict__ Y, const float* __restrict__ gamma,
    const float* __restrict__ beta) {
  __shared__ float red[8];
  float*    p = Y + (size_t)blockIdx.x * HH;
  const int t = threadIdx.x;

  float x[4];
#pragma unroll
  for (int i = 0; i < 4; ++i) x[i] = p[t + 256 * i];

  float s = x[0] + x[1] + x[2] + x[3];
#pragma unroll
  for (int off = 16; off > 0; off >>= 1) s += __shfl_xor(s, off, 32);
  if ((t & 31) == 0) red[t >> 5] = s;
  __syncthreads();
  float tot = 0.0f;
#pragma unroll
  for (int i = 0; i < 8; ++i) tot += red[i];
  const float mu = tot * (1.0f / (float)HH);
  __syncthreads();

  float vs = 0.0f;
#pragma unroll
  for (int i = 0; i < 4; ++i) { const float d = x[i] - mu; vs += d * d; }
#pragma unroll
  for (int off = 16; off > 0; off >>= 1) vs += __shfl_xor(vs, off, 32);
  if ((t & 31) == 0) red[t >> 5] = vs;
  __syncthreads();
  float vtot = 0.0f;
#pragma unroll
  for (int i = 0; i < 8; ++i) vtot += red[i];
  const float rstd = rsqrtf(vtot * (1.0f / (float)HH) + 1e-12f);

#pragma unroll
  for (int i = 0; i < 4; ++i) {
    const int c = t + 256 * i;
    p[c] = (x[i] - mu) * rstd * gamma[c] + beta[c];
  }
}

// ---------------------------------------------------------------------------
// launch
// ---------------------------------------------------------------------------
extern "C" void kernel_launch(void* const* d_in, const int* in_sizes, int n_in,
                              void* d_out, int out_size, void* d_ws,
                              size_t ws_size, hipStream_t stream) {
  (void)in_sizes; (void)n_in; (void)out_size; (void)ws_size;
  const float* ini_q = (const float*)d_in[0];
  const float* ini_k = (const float*)d_in[1];
  const float* mask  = (const float*)d_in[2];
  const float* Wq    = (const float*)d_in[3];
  const float* bq    = (const float*)d_in[4];
  const float* Wk    = (const float*)d_in[5];
  const float* bk    = (const float*)d_in[6];
  const float* Wv    = (const float*)d_in[7];
  const float* bv    = (const float*)d_in[8];
  const float* Wo    = (const float*)d_in[9];
  const float* bo    = (const float*)d_in[10];
  const float* gamma = (const float*)d_in[11];
  const float* beta  = (const float*)d_in[12];
  float*       out   = (float*)d_out;

  const size_t NKV  = (size_t)NB * NS * NT;  // 65536 rows
  const size_t NRES = (size_t)NB * NQ * NS;  // 16384 rows

  // ws layout: [k16 128MB][v16 128MB][res f32 64MB][q16 0.5MB] ~ 320.5 MB
  _Float16* k16 = (_Float16*)d_ws;
  _Float16* v16 = k16 + NKV * HH;
  float*    res = (float*)(v16 + NKV * HH);
  _Float16* q16 = (_Float16*)(res + NRES * HH);

  dim3 blk(128);
  hipLaunchKernelGGL((gemm_bias_kernel<_Float16>), dim3(16, 4), blk, 0, stream,
                     ini_q, Wq, bq, q16);
  hipLaunchKernelGGL(gemm_kv_kernel, dim3(16, 1024), blk, 0, stream,
                     ini_k, Wk, bk, Wv, bv, k16, v16);
  hipLaunchKernelGGL(attn_kernel, dim3(NB * NS), dim3(128), 0, stream, q16,
                     k16, v16, mask, res);
  hipLaunchKernelGGL((gemm_bias_kernel<float>), dim3(16, 256), blk, 0, stream,
                     res, Wo, bo, out);
  hipLaunchKernelGGL(layernorm_inplace, dim3((unsigned)NRES), dim3(256), 0,
                     stream, out, gamma, beta);
}
